// HeteroGraphGAT_25125558681999
// MI455X (gfx1250) — matrix operand order repaired and checked by reading, not compiled, not run on hardware
//
#include <hip/hip_runtime.h>

// ---------------------------------------------------------------------------
// HeteroGraph GAT (2 node types, 2 edge types, L=2) for MI455X / gfx1250.
//
// Roofline: dense GEMMs are only ~26 GFLOP total -> done with
// v_wmma_f32_16x16x32_bf16 (fp32 accumulate). Dominant cost is the edge-wise
// gather/scatter (~3.3 GB irregular traffic); dst accumulators (25.6 MB)
// fit in the 192 MB L2 so atomic f32 adds stay on-chip. The scatter uses
// hardware global_atomic_add_f32 at device scope via inline asm (guaranteed
// no CAS expansion). The E x DE x C edge GEMM is folded into a tiny [H][DE]
// projection since its output is only contracted with a_e.
// ---------------------------------------------------------------------------

#define NNODES 50000
#define CDIM   128
#define HEADS  4
#define CHD    32          // channels per head
#define EDGES  800000
#define DEDIM  16
#define LAYERS 2
#define NEG_SLOPE 0.2f
#define LNEPS  1e-5f

typedef __attribute__((ext_vector_type(16))) __bf16    v16bf;
typedef __attribute__((ext_vector_type(8)))  float     v8f;
typedef __attribute__((ext_vector_type(4)))  unsigned  u32x4;

// ----------------------------- device helpers ------------------------------

__device__ __forceinline__ __bf16 f2bf(float f) {
  unsigned b = __float_as_uint(f);
  unsigned r = b + 0x7FFFu + ((b >> 16) & 1u);          // round-to-nearest-even
  union { unsigned short u; __bf16 b; } cv;
  cv.u = (unsigned short)(r >> 16);
  return cv.b;
}

// order-preserving float<->uint mapping for atomicMax over signed floats
__device__ __forceinline__ unsigned fkey(float f) {
  unsigned b = __float_as_uint(f);
  return (b & 0x80000000u) ? ~b : (b | 0x80000000u);
}
__device__ __forceinline__ float funkey(unsigned k) {
  unsigned b = (k & 0x80000000u) ? (k & 0x7FFFFFFFu) : ~k;
  return __uint_as_float(b);
}

// Hardware f32 atomic add, device scope (lands in GL2). Inline asm so the
// backend can never CAS-expand the hot scatter path.
__device__ __forceinline__ void atomicAddF(float* p, float v) {
  asm volatile("global_atomic_add_f32 %0, %1, off scope:SCOPE_DEV"
               :
               : "v"(p), "v"(v)
               : "memory");
}

__device__ __forceinline__ float dot4(float4 a, float4 b) {
  return a.x * b.x + a.y * b.y + a.z * b.z + a.w * b.w;
}

// ------------------------------- kernels -----------------------------------

__global__ void k_f32_to_bf16(const float* __restrict__ in,
                              __bf16* __restrict__ out, int n) {
  int t = blockIdx.x * blockDim.x + threadIdx.x;
  if (t < n) out[t] = f2bf(in[t]);
}

// Wt[nOut][kIn] = bf16(W[kIn][nOut]); W is [CDIM][CDIM] fp32
__global__ void k_transpose_w(const float* __restrict__ W,
                              __bf16* __restrict__ Wt) {
  int t = blockIdx.x * blockDim.x + threadIdx.x;      // 16384 threads
  int nO = t >> 7, kI = t & 127;
  Wt[nO * CDIM + kI] = f2bf(W[kI * CDIM + nO]);
}

__global__ void k_zero(float* __restrict__ p, int n) {
  int t = blockIdx.x * blockDim.x + threadIdx.x;
  if (t < n) p[t] = 0.f;
}

// Out[M,C] = A[M,C](bf16) @ Wt^T  where Wt is [C_out][C_in] bf16 row-major.
// One wave per 16x16 tile; K-loop = 4 x wmma_f32_16x16x32_bf16.
__global__ __launch_bounds__(256) void k_gemm_bf16(
    const __bf16* __restrict__ A, const __bf16* __restrict__ Wt,
    float* __restrict__ Out) {
  const int lane  = threadIdx.x & 31;
  const int wv    = threadIdx.x >> 5;
  const int tile  = blockIdx.x * 8 + wv;      // 25000 tiles total
  const int mbase = (tile >> 3) * 16;         // 3125 M-tiles
  const int nbase = (tile & 7) * 16;          // 8 N-tiles
  const int half  = lane >> 4;
  const int l16   = lane & 15;

  const __bf16* arow = A  + (size_t)(mbase + l16) * CDIM;
  const __bf16* brow = Wt + (size_t)(nbase + l16) * CDIM;

  v8f acc = {};
#pragma unroll
  for (int k0 = 0; k0 < CDIM; k0 += 32) {
    union { u32x4 u[2]; v16bf v; } af, bfm;
    // A lane layout (16-bit 16x32): lanes 0-15 hold K=[k0..k0+7, k0+16..k0+23]
    //                               lanes 16-31 hold K=[k0+8..+15, k0+24..+31]
    const int ka = k0 + half * 8;
    af.u[0] = *(const u32x4*)(arow + ka);
    af.u[1] = *(const u32x4*)(arow + ka + 16);
    // B lane layout (32x16): lanes 0-15: N=lane, K=[k0..k0+7, k0+16..k0+23];
    //                        lanes 16-31: N=lane-16, K=[k0+8..+15, k0+24..+31]
    const int kb = k0 + half * 8;
    bfm.u[0] = *(const u32x4*)(brow + kb);
    bfm.u[1] = *(const u32x4*)(brow + kb + 16);
    acc = __builtin_amdgcn_wmma_f32_16x16x32_bf16(false, af.v, false, bfm.v,
                                                  (short)0, acc, false, false);
  }
  // D layout: VGPR r -> M = r + 8*half, N = l16
  float* orow = Out + (size_t)(mbase + 8 * half) * CDIM + nbase + l16;
#pragma unroll
  for (int r = 0; r < 8; ++r) orow[(size_t)r * CDIM] = acc[r];
}

// al[n,h] = sum_c X[n, h*32+c] * avec[h*32+c]
__global__ void k_attn_node(const float* __restrict__ X,
                            const float* __restrict__ avec,
                            float* __restrict__ al, int total) {
  int t = blockIdx.x * blockDim.x + threadIdx.x;
  if (t >= total) return;
  int n = t >> 2, h = t & 3;
  const float4* xp = (const float4*)(X + (size_t)n * CDIM + h * CHD);
  const float4* ap = (const float4*)(avec + h * CHD);
  float s = 0.f;
#pragma unroll
  for (int i = 0; i < 8; ++i) s += dot4(xp[i], ap[i]);
  al[t] = s;
}

// ve[h][de] = sum_c We[de, h*32+c] * ae[h*32+c]   (folds the edge GEMM)
__global__ void k_ve(const float* __restrict__ We, const float* __restrict__ ae,
                     float* __restrict__ ve) {
  int t = threadIdx.x;
  if (t >= HEADS * DEDIM) return;
  int h = t >> 4, de = t & 15;
  const float* w = We + (size_t)de * CDIM + h * CHD;
  const float* a = ae + h * CHD;
  float s = 0.f;
  for (int c = 0; c < CHD; ++c) s += w[c] * a[c];
  ve[h * DEDIM + de] = s;
}

__global__ void k_init_ms(unsigned* __restrict__ mkey, float* __restrict__ ssum,
                          int total) {
  int t = blockIdx.x * blockDim.x + threadIdx.x;
  if (t >= total) return;
  mkey[t] = 0x007FFFFFu;  // fkey(-inf)
  ssum[t] = 0.f;
}

// alpha[e,h] = leaky_relu(al_s[src,h] + al_d[dst,h] + eattr[e].ve[h]);
// atomicMax running segment max per (dst,h)
__global__ void k_edge_alpha(const int* __restrict__ ei,
                             const float* __restrict__ eattr,
                             const float* __restrict__ ve,
                             const float* __restrict__ als,
                             const float* __restrict__ ald,
                             float* __restrict__ alpha,
                             unsigned* __restrict__ mkey) {
  int e = blockIdx.x * blockDim.x + threadIdx.x;
  if (e >= EDGES) return;
  int src = ei[e], dst = ei[EDGES + e];
  const float4* ep = (const float4*)(eattr + (size_t)e * DEDIM);
  float4 e0 = ep[0], e1 = ep[1], e2 = ep[2], e3 = ep[3];
  float4 out;
  float* po = &out.x;
#pragma unroll
  for (int h = 0; h < 4; ++h) {
    const float4* vp = (const float4*)(ve + h * DEDIM);
    float ale = dot4(e0, vp[0]) + dot4(e1, vp[1]) + dot4(e2, vp[2]) +
                dot4(e3, vp[3]);
    float a = als[src * 4 + h] + ald[dst * 4 + h] + ale;
    a = (a > 0.f) ? a : NEG_SLOPE * a;
    po[h] = a;
    atomicMax(&mkey[dst * 4 + h], fkey(a));
  }
  *(float4*)(alpha + (size_t)e * 4) = out;
}

__global__ void k_edge_expsum(const int* __restrict__ ei,
                              const float* __restrict__ alpha,
                              const unsigned* __restrict__ mkey,
                              float* __restrict__ ssum) {
  int e = blockIdx.x * blockDim.x + threadIdx.x;
  if (e >= EDGES) return;
  int dst = ei[EDGES + e];
  float4 a = *(const float4*)(alpha + (size_t)e * 4);
  const float* pa = &a.x;
#pragma unroll
  for (int h = 0; h < 4; ++h) {
    float m = funkey(mkey[dst * 4 + h]);
    atomicAddF(&ssum[dst * 4 + h], __expf(pa[h] - m));
  }
}

// acc[dst, c..c+3] += xs[src, c..c+3] * softmax_weight(e, h(c)).
// 32 threads (one wave) per edge, 4 channels each; weight recomputed
// (cheaper than storing/reloading E*H ex values).
__global__ __launch_bounds__(256) void k_message(
    const int* __restrict__ ei, const float* __restrict__ alpha,
    const unsigned* __restrict__ mkey, const float* __restrict__ ssum,
    const float* __restrict__ xs, float* __restrict__ acc) {
  unsigned t = blockIdx.x * 256u + threadIdx.x;  // EDGES*32 threads exactly
  int e = t >> 5;
  int q = t & 31;
  int h = q >> 3;
  int c = q << 2;
  int src = ei[e], dst = ei[EDGES + e];
  float a = alpha[(size_t)e * 4 + h];
  float m = funkey(mkey[dst * 4 + h]);
  float s = ssum[dst * 4 + h];
  float w = __expf(a - m) / (s + 1e-16f);
  float4 v = *(const float4*)(xs + (size_t)src * CDIM + c);
  float* o = acc + (size_t)dst * CDIM + c;
  atomicAddF(o + 0, v.x * w);
  atomicAddF(o + 1, v.y * w);
  atomicAddF(o + 2, v.z * w);
  atomicAddF(o + 3, v.w * w);
}

// One wave32 per node: y = relu(LN(acc + bias) * g + b); write bf16 features
// for next layer and (last layer) fp32 into d_out.
__global__ __launch_bounds__(256) void k_epilogue(
    const float* __restrict__ acc, const float* __restrict__ bias,
    const float* __restrict__ g, const float* __restrict__ bb,
    __bf16* __restrict__ xbf, float* __restrict__ outF, int writeOut) {
  int lane = threadIdx.x & 31;
  int wv = threadIdx.x >> 5;
  int n = blockIdx.x * 8 + wv;
  int c = lane * 4;
  float4 a = *(const float4*)(acc + (size_t)n * CDIM + c);
  float4 bi = *(const float4*)(bias + c);
  float vx = a.x + bi.x, vy = a.y + bi.y, vz = a.z + bi.z, vw = a.w + bi.w;
  float s = vx + vy + vz + vw;
#pragma unroll
  for (int m = 16; m >= 1; m >>= 1) s += __shfl_xor(s, m, 32);
  float mu = s * (1.f / 128.f);
  float dx = vx - mu, dy = vy - mu, dz = vz - mu, dw = vw - mu;
  float q = dx * dx + dy * dy + dz * dz + dw * dw;
#pragma unroll
  for (int m = 16; m >= 1; m >>= 1) q += __shfl_xor(q, m, 32);
  float rs = rsqrtf(q * (1.f / 128.f) + LNEPS);
  float4 gg = *(const float4*)(g + c);
  float4 b2 = *(const float4*)(bb + c);
  float y0 = fmaxf(dx * rs * gg.x + b2.x, 0.f);
  float y1 = fmaxf(dy * rs * gg.y + b2.y, 0.f);
  float y2 = fmaxf(dz * rs * gg.z + b2.z, 0.f);
  float y3 = fmaxf(dw * rs * gg.w + b2.w, 0.f);
  __bf16* ob = xbf + (size_t)n * CDIM + c;
  ob[0] = f2bf(y0); ob[1] = f2bf(y1); ob[2] = f2bf(y2); ob[3] = f2bf(y3);
  if (writeOut) {
    *(float4*)(outF + (size_t)n * CDIM + c) = make_float4(y0, y1, y2, y3);
  }
}

// ------------------------------ host driver --------------------------------

extern "C" void kernel_launch(void* const* d_in, const int* in_sizes, int n_in,
                              void* d_out, int out_size, void* d_ws,
                              size_t ws_size, hipStream_t stream) {
  (void)in_sizes; (void)n_in; (void)out_size; (void)ws_size;
  const float* x_user   = (const float*)d_in[0];
  const float* x_item   = (const float*)d_in[1];
  const float* eattr_ui = (const float*)d_in[2];
  const float* eattr_iu = (const float*)d_in[3];
  const float* Wsrc_ui  = (const float*)d_in[4];
  const float* Wdst_ui  = (const float*)d_in[5];
  const float* We_ui    = (const float*)d_in[6];
  const float* asrc_ui  = (const float*)d_in[7];
  const float* adst_ui  = (const float*)d_in[8];
  const float* ae_ui    = (const float*)d_in[9];
  const float* b_ui     = (const float*)d_in[10];
  const float* Wsrc_iu  = (const float*)d_in[11];
  const float* Wdst_iu  = (const float*)d_in[12];
  const float* We_iu    = (const float*)d_in[13];
  const float* asrc_iu  = (const float*)d_in[14];
  const float* adst_iu  = (const float*)d_in[15];
  const float* ae_iu    = (const float*)d_in[16];
  const float* b_iu     = (const float*)d_in[17];
  const float* ln_u_g   = (const float*)d_in[18];
  const float* ln_u_b   = (const float*)d_in[19];
  const float* ln_i_g   = (const float*)d_in[20];
  const float* ln_i_b   = (const float*)d_in[21];
  const int*   ei_ui    = (const int*)d_in[22];
  const int*   ei_iu    = (const int*)d_in[23];
  float* outF = (float*)d_out;

  char* base = (char*)d_ws;
  size_t off = 0;
  auto alloc = [&](size_t bytes) -> void* {
    void* p = base + off;
    off += (bytes + 255) & ~(size_t)255;
    return p;
  };
  __bf16* xbf_u = (__bf16*)alloc((size_t)NNODES * CDIM * 2);
  __bf16* xbf_i = (__bf16*)alloc((size_t)NNODES * CDIM * 2);
  __bf16* wt_s  = (__bf16*)alloc((size_t)CDIM * CDIM * 2);
  __bf16* wt_d  = (__bf16*)alloc((size_t)CDIM * CDIM * 2);
  float* xs    = (float*)alloc((size_t)NNODES * CDIM * 4);
  float* xd    = (float*)alloc((size_t)NNODES * CDIM * 4);
  float* al_s  = (float*)alloc((size_t)NNODES * HEADS * 4);
  float* al_d  = (float*)alloc((size_t)NNODES * HEADS * 4);
  float* ve    = (float*)alloc((size_t)HEADS * DEDIM * 4);
  float* alpha = (float*)alloc((size_t)EDGES * HEADS * 4);
  unsigned* mkey = (unsigned*)alloc((size_t)NNODES * HEADS * 4);
  float* ssum  = (float*)alloc((size_t)NNODES * HEADS * 4);
  float* acc_i = (float*)alloc((size_t)NNODES * CDIM * 4);
  float* acc_u = (float*)alloc((size_t)NNODES * CDIM * 4);

  const dim3 B(256);
  const int NH = NNODES * HEADS;

  k_f32_to_bf16<<<NNODES * CDIM / 256, B, 0, stream>>>(x_user, xbf_u,
                                                       NNODES * CDIM);
  k_f32_to_bf16<<<NNODES * CDIM / 256, B, 0, stream>>>(x_item, xbf_i,
                                                       NNODES * CDIM);

  auto run_conv = [&](const __bf16* srcF, const __bf16* dstF, const int* ei,
                      const float* eattr, const float* Ws, const float* Wd,
                      const float* We, const float* as_, const float* ad_,
                      const float* ae_, float* acc) {
    k_transpose_w<<<64, B, 0, stream>>>(Ws, wt_s);
    k_transpose_w<<<64, B, 0, stream>>>(Wd, wt_d);
    k_gemm_bf16<<<3125, B, 0, stream>>>(srcF, wt_s, xs);
    k_gemm_bf16<<<3125, B, 0, stream>>>(dstF, wt_d, xd);
    k_attn_node<<<(NH + 255) / 256, B, 0, stream>>>(xs, as_, al_s, NH);
    k_attn_node<<<(NH + 255) / 256, B, 0, stream>>>(xd, ad_, al_d, NH);
    k_ve<<<1, 64, 0, stream>>>(We, ae_, ve);
    k_init_ms<<<(NH + 255) / 256, B, 0, stream>>>(mkey, ssum, NH);
    k_zero<<<NNODES * CDIM / 256, B, 0, stream>>>(acc, NNODES * CDIM);
    k_edge_alpha<<<EDGES / 256, B, 0, stream>>>(ei, eattr, ve, al_s, al_d,
                                                alpha, mkey);
    k_edge_expsum<<<EDGES / 256, B, 0, stream>>>(ei, alpha, mkey, ssum);
    k_message<<<EDGES * 32 / 256, B, 0, stream>>>(ei, alpha, mkey, ssum, xs,
                                                  acc);
  };

  for (int l = 0; l < LAYERS; ++l) {
    size_t oCC = (size_t)l * CDIM * CDIM;
    size_t oDC = (size_t)l * DEDIM * CDIM;
    size_t oHC = (size_t)l * HEADS * CHD;
    size_t oC  = (size_t)l * CDIM;
    // user -> item (dst = item)
    run_conv(xbf_u, xbf_i, ei_ui, eattr_ui, Wsrc_ui + oCC, Wdst_ui + oCC,
             We_ui + oDC, asrc_ui + oHC, adst_ui + oHC, ae_ui + oHC, acc_i);
    // item -> user (dst = user)
    run_conv(xbf_i, xbf_u, ei_iu, eattr_iu, Wsrc_iu + oCC, Wdst_iu + oCC,
             We_iu + oDC, asrc_iu + oHC, adst_iu + oHC, ae_iu + oHC, acc_u);
    int last = (l == LAYERS - 1);
    k_epilogue<<<NNODES / 8, B, 0, stream>>>(acc_u, b_iu + oC, ln_u_g + oC,
                                             ln_u_b + oC, xbf_u, outF, last);
    k_epilogue<<<NNODES / 8, B, 0, stream>>>(acc_i, b_ui + oC, ln_i_g + oC,
                                             ln_i_b + oC, xbf_i,
                                             outF + (size_t)NNODES * CDIM,
                                             last);
  }
}
